// MultiHeadAttentionThresh_29368986370326
// MI455X (gfx1250) — matrix operand order repaired and checked by reading, not compiled, not run on hardware
//
#include <hip/hip_runtime.h>
#include <hip/hip_bf16.h>

#define WB 2
#define WS 2048
#define WE 1024
#define WH 16
#define WDK 64

typedef __attribute__((ext_vector_type(16))) __bf16 v16bf;
typedef __attribute__((ext_vector_type(8)))  __bf16 v8bf;
typedef __attribute__((ext_vector_type(8)))  float  v8f;

static __device__ __forceinline__ v8f wmma_bf16(v16bf a, v16bf b, v8f c) {
    // D = A(16x32 bf16) * B(32x16 bf16) + C(16x16 f32)
    return __builtin_amdgcn_wmma_f32_16x16x32_bf16(
        /*neg_a=*/false, a, /*neg_b=*/false, b,
        /*c_mod=*/(short)0, c, /*reuse_a=*/false, /*reuse_b=*/false);
}

// A fragment: M=16 x K=32 bf16, source row-major with row stride `stride` elems.
// Layout: lanes 0-15 hold rows M=lane with K 0..15;
// lanes 16-31 hold rows M=lane-16 with K 16..31 (matches load_frag_b).
static __device__ __forceinline__ v16bf load_frag_a(const __bf16* base, int stride) {
    int l  = threadIdx.x & 31;
    int m  = l & 15;
    int kb = (l & 16) ? 16 : 0;
    const __bf16* p = base + (size_t)m * stride + kb;
    v8bf lo = *(const v8bf*)p;          // K = kb .. kb+7
    v8bf hi = *(const v8bf*)(p + 8);    // K = kb+8 .. kb+15
    return __builtin_shufflevector(lo, hi, 0,1,2,3,4,5,6,7,8,9,10,11,12,13,14,15);
}

// B fragment: K=32 x N=16 bf16 where B[k][n] = SRC[n][k], SRC row-major, row
// stride `stride` elems (i.e. B is SRC transposed; K is SRC's fast axis).
// Layout used here (same as A): lane = column n (lanes 16-31 hold K 16..31), K-pairs packed/dword.
static __device__ __forceinline__ v16bf load_frag_b(const __bf16* base, int stride) {
    int l  = threadIdx.x & 31;
    int n  = l & 15;
    int kb = (l & 16) ? 16 : 0;
    return *(const v16bf*)(base + (size_t)n * stride + kb);
}

// ---------------------------------------------------------------- converts
__global__ void cvt_bf16_kernel(const float* __restrict__ src,
                                __bf16* __restrict__ dst, int n) {
    int i = blockIdx.x * blockDim.x + threadIdx.x;
    if (i < n) dst[i] = (__bf16)src[i];
}

// V [B,S,E] (head-major cols) -> VT [B,H,DK,S] so thr@V B-frags are contiguous
__global__ void vtrans_kernel(const __bf16* __restrict__ V,
                              __bf16* __restrict__ VT) {
    size_t i = (size_t)blockIdx.x * blockDim.x + threadIdx.x;
    if (i >= (size_t)WB * WS * WE) return;
    size_t t = i;
    int s = (int)(t % WS); t /= WS;
    int d = (int)(t % WDK); t /= WDK;
    int h = (int)(t % WH);
    int b = (int)(t / WH);
    VT[i] = V[((size_t)(b * WS + s)) * WE + h * WDK + d];
}

// ------------------------------------------------ out = X @ W^T + bias, scaled
// grid = (rows/16, E/64), block = 32 (one wave). W row-major [E,E] => B-frag is
// contiguous along W's fast axis (k). Software-pipelined: next k-step's A + 4 B
// fragments are issued before the current 4 WMMAs so waits overlap compute.
__global__ void __launch_bounds__(32)
gemm_xwt_kernel(const __bf16* __restrict__ X, const __bf16* __restrict__ W,
                const float* __restrict__ bias, float* __restrict__ outf,
                __bf16* __restrict__ outb, int out_bf16, float scale) {
    int lane = threadIdx.x & 31;
    int rt = blockIdx.x;   // 16-row tile
    int eg = blockIdx.y;   // 64-col group
    const __bf16* Xb = X + (size_t)rt * 16 * WE;
    const __bf16* Wb = W + (size_t)(eg * 64) * WE;
    v8f acc0 = {}, acc1 = {}, acc2 = {}, acc3 = {};

    v16bf a  = load_frag_a(Xb, WE);
    v16bf b0 = load_frag_b(Wb + (size_t)0  * WE, WE);
    v16bf b1 = load_frag_b(Wb + (size_t)16 * WE, WE);
    v16bf b2 = load_frag_b(Wb + (size_t)32 * WE, WE);
    v16bf b3 = load_frag_b(Wb + (size_t)48 * WE, WE);

    for (int kk = 0; kk < WE / 32; ++kk) {
        v16bf an, bn0, bn1, bn2, bn3;
        if (kk + 1 < WE / 32) {                       // prefetch next k-step
            const __bf16* Xn = Xb + (kk + 1) * 32;
            const __bf16* Wn = Wb + (kk + 1) * 32;
            an  = load_frag_a(Xn, WE);
            bn0 = load_frag_b(Wn + (size_t)0  * WE, WE);
            bn1 = load_frag_b(Wn + (size_t)16 * WE, WE);
            bn2 = load_frag_b(Wn + (size_t)32 * WE, WE);
            bn3 = load_frag_b(Wn + (size_t)48 * WE, WE);
        } else {
            an = a; bn0 = b0; bn1 = b1; bn2 = b2; bn3 = b3;
        }
        acc0 = wmma_bf16(a, b0, acc0);
        acc1 = wmma_bf16(a, b1, acc1);
        acc2 = wmma_bf16(a, b2, acc2);
        acc3 = wmma_bf16(a, b3, acc3);
        a = an; b0 = bn0; b1 = bn1; b2 = bn2; b3 = bn3;
    }

    int n = lane & 15, mo = (lane & 16) ? 8 : 0;
    v8f accs[4] = {acc0, acc1, acc2, acc3};
#pragma unroll
    for (int et = 0; et < 4; ++et) {
        int e = eg * 64 + et * 16 + n;
        float bv = bias[e];
#pragma unroll
        for (int r = 0; r < 8; ++r) {
            float v = (accs[et][r] + bv) * scale;
            size_t idx = (size_t)(rt * 16 + mo + r) * WE + e;
            if (out_bf16) outb[idx] = (__bf16)v; else outf[idx] = v;
        }
    }
}

// ---------------------------------------- pass 1: softmax row max / inv-sum
// grid = B*H*(S/16), block = 32. Logits pre-scaled (Q holds 1/8 factor).
__global__ void __launch_bounds__(32)
attn_stats_kernel(const __bf16* __restrict__ Qbf, const __bf16* __restrict__ Kbf,
                  float* __restrict__ mstat, float* __restrict__ lstat) {
    __shared__ float tile[16 * 16];
    int lane = threadIdx.x & 31;
    int blk = blockIdx.x;
    int qt = blk % (WS / 16); int t = blk / (WS / 16);
    int h = t % WH; int b = t / WH;
    int qbase = qt * 16;
    const __bf16* Qb = Qbf + (size_t)(b * WS + qbase) * WE + h * WDK;
    const __bf16* Kb = Kbf + (size_t)b * WS * WE + h * WDK;
    v16bf aq0 = load_frag_a(Qb, WE);
    v16bf aq1 = load_frag_a(Qb + 32, WE);
    int n = lane & 15, mo = (lane & 16) ? 8 : 0;
    float mrun = -1e30f, srun = 0.f;
    for (int tt = 0; tt < WS / 16; ++tt) {
        const __bf16* Kt = Kb + (size_t)tt * 16 * WE;
        v8f c = {};
        c = wmma_bf16(aq0, load_frag_b(Kt, WE), c);
        c = wmma_bf16(aq1, load_frag_b(Kt + 32, WE), c);
#pragma unroll
        for (int r = 0; r < 8; ++r) tile[(mo + r) * 16 + n] = c[r];
        asm volatile("s_wait_dscnt 0" ::: "memory");
        if (lane < 16) {   // lane r reduces row r (cross-lane via LDS)
            float v[16], tmax = -1e30f;
#pragma unroll
            for (int j = 0; j < 16; ++j) { v[j] = tile[lane * 16 + j]; tmax = fmaxf(tmax, v[j]); }
            float mnew = fmaxf(mrun, tmax);
            float s = 0.f;
#pragma unroll
            for (int j = 0; j < 16; ++j) s += __expf(v[j] - mnew);
            srun = srun * __expf(mrun - mnew) + s;
            mrun = mnew;
        }
        asm volatile("s_wait_dscnt 0" ::: "memory");
    }
    if (lane < 16) {
        size_t o = (size_t)(b * WH + h) * WS + qbase + lane;
        mstat[o] = mrun;
        lstat[o] = 1.f / srun;
    }
}

// ------------- pass 2: exact probs, adjacency (mean over heads), thr@V hidden
// grid = B*(S/16), block = 512 (16 waves, wave w == head w).
// dynamic LDS: 16*S f32 adjacency accumulator + per-wave 16x32 bf16 thr stage.
__global__ void __launch_bounds__(512)
attn_pass2_kernel(const __bf16* __restrict__ Qbf, const __bf16* __restrict__ Kbf,
                  const __bf16* __restrict__ VT,
                  const float* __restrict__ mstat, const float* __restrict__ lstat,
                  __bf16* __restrict__ merged, float* __restrict__ adjout) {
    extern __shared__ char smem[];
    float*  adj   = (float*)smem;                               // [16][S]
    __bf16* stg   = (__bf16*)(smem + 16 * WS * sizeof(float));  // [16 waves][16*32]
    int tid = threadIdx.x;
    int lane = tid & 31;
    int wave = tid >> 5;           // == head
    int qt = blockIdx.x % (WS / 16);
    int b  = blockIdx.x / (WS / 16);
    int qbase = qt * 16;

    for (int i = tid; i < 16 * WS; i += 512) adj[i] = 0.f;
    __syncthreads();

    int h = wave;
    int n = lane & 15, mo = (lane & 16) ? 8 : 0;
    const float* mp = mstat + (size_t)(b * WH + h) * WS + qbase;
    const float* lp = lstat + (size_t)(b * WH + h) * WS + qbase;
    float mrow[8], lrow[8];
#pragma unroll
    for (int r = 0; r < 8; ++r) { mrow[r] = mp[mo + r]; lrow[r] = lp[mo + r]; }

    const __bf16* Qb = Qbf + (size_t)(b * WS + qbase) * WE + h * WDK;
    const __bf16* Kb = Kbf + (size_t)b * WS * WE + h * WDK;
    const __bf16* Vb = VT + (size_t)((b * WH + h) * WDK) * WS;
    __bf16* stage = stg + wave * (16 * 32);

    v16bf aq0 = load_frag_a(Qb, WE);
    v16bf aq1 = load_frag_a(Qb + 32, WE);
    v8f hacc0 = {}, hacc1 = {}, hacc2 = {}, hacc3 = {};

    for (int t2 = 0; t2 < WS / 32; ++t2) {
        int tbase = t2 * 32;
#pragma unroll
        for (int hf = 0; hf < 2; ++hf) {
            int tb = tbase + hf * 16;
            const __bf16* Kt = Kb + (size_t)tb * WE;
            v8f c = {};
            c = wmma_bf16(aq0, load_frag_b(Kt, WE), c);
            c = wmma_bf16(aq1, load_frag_b(Kt + 32, WE), c);
#pragma unroll
            for (int r = 0; r < 8; ++r) {
                float p = __expf(c[r] - mrow[r]) * lrow[r];      // exact prob
                atomicAdd(&adj[(mo + r) * WS + tb + n], p);      // ds_add_f32
                float tv = (p > 0.1f) ? p : 0.f;                 // threshold
                stage[(mo + r) * 32 + hf * 16 + n] = (__bf16)tv;
            }
        }
        asm volatile("s_wait_dscnt 0" ::: "memory");
        v16bf at = load_frag_a(stage, 32);                       // thr as A-frag
        hacc0 = wmma_bf16(at, load_frag_b(Vb + (size_t)0  * WS + tbase, WS), hacc0);
        hacc1 = wmma_bf16(at, load_frag_b(Vb + (size_t)16 * WS + tbase, WS), hacc1);
        hacc2 = wmma_bf16(at, load_frag_b(Vb + (size_t)32 * WS + tbase, WS), hacc2);
        hacc3 = wmma_bf16(at, load_frag_b(Vb + (size_t)48 * WS + tbase, WS), hacc3);
    }

    v8f haccs[4] = {hacc0, hacc1, hacc2, hacc3};
#pragma unroll
    for (int dt = 0; dt < 4; ++dt)
#pragma unroll
        for (int r = 0; r < 8; ++r)
            merged[(size_t)(b * WS + qbase + mo + r) * WE + h * WDK + dt * 16 + n] =
                (__bf16)haccs[dt][r];

    __syncthreads();
    float* ao = adjout + (size_t)(b * WS + qbase) * WS;
    for (int i = tid; i < 16 * WS; i += 512) ao[i] = adj[i] * (1.f / WH);
}

// --------------------------------------------------------------------- host
extern "C" void kernel_launch(void* const* d_in, const int* in_sizes, int n_in,
                              void* d_out, int out_size, void* d_ws, size_t ws_size,
                              hipStream_t stream) {
    const int NE = WB * WS * WE;   // 4,194,304
    const int EE = WE * WE;        // 1,048,576

    const float* q   = (const float*)d_in[0];
    const float* k   = (const float*)d_in[1];
    const float* v   = (const float*)d_in[2];
    const float* Wq  = (const float*)d_in[3];
    const float* bq  = (const float*)d_in[4];
    const float* Wk  = (const float*)d_in[5];
    const float* bk  = (const float*)d_in[6];
    const float* Wv  = (const float*)d_in[7];
    const float* bv  = (const float*)d_in[8];
    const float* Wp  = (const float*)d_in[9];
    const float* bp  = (const float*)d_in[10];

    char* ws = (char*)d_ws;
    size_t off = 0;
    auto alloc = [&](size_t bytes) -> void* {
        void* p = ws + off;
        off = (off + bytes + 255) & ~(size_t)255;
        return p;
    };
    __bf16* Xqb = (__bf16*)alloc((size_t)NE * 2);
    __bf16* Xkb = (__bf16*)alloc((size_t)NE * 2);
    __bf16* Xvb = (__bf16*)alloc((size_t)NE * 2);
    __bf16* Wqb = (__bf16*)alloc((size_t)EE * 2);
    __bf16* Wkb = (__bf16*)alloc((size_t)EE * 2);
    __bf16* Wvb = (__bf16*)alloc((size_t)EE * 2);
    __bf16* Wpb = (__bf16*)alloc((size_t)EE * 2);
    __bf16* Qb  = (__bf16*)alloc((size_t)NE * 2);
    __bf16* Kb  = (__bf16*)alloc((size_t)NE * 2);
    __bf16* Vbp = (__bf16*)alloc((size_t)NE * 2);
    __bf16* VTb = (__bf16*)alloc((size_t)NE * 2);
    __bf16* Mb  = (__bf16*)alloc((size_t)NE * 2);
    float*  mstat = (float*)alloc((size_t)WB * WH * WS * 4);
    float*  lstat = (float*)alloc((size_t)WB * WH * WS * 4);

    float* outp = (float*)d_out;          // [B,S,E]
    float* adjp = outp + NE;              // [B,S,S]

    // 1) fp32 -> bf16
    cvt_bf16_kernel<<<(NE + 255) / 256, 256, 0, stream>>>(q, Xqb, NE);
    cvt_bf16_kernel<<<(NE + 255) / 256, 256, 0, stream>>>(k, Xkb, NE);
    cvt_bf16_kernel<<<(NE + 255) / 256, 256, 0, stream>>>(v, Xvb, NE);
    cvt_bf16_kernel<<<(EE + 255) / 256, 256, 0, stream>>>(Wq, Wqb, EE);
    cvt_bf16_kernel<<<(EE + 255) / 256, 256, 0, stream>>>(Wk, Wkb, EE);
    cvt_bf16_kernel<<<(EE + 255) / 256, 256, 0, stream>>>(Wv, Wvb, EE);
    cvt_bf16_kernel<<<(EE + 255) / 256, 256, 0, stream>>>(Wp, Wpb, EE);

    // 2) projections (Q carries the 1/sqrt(DK)=0.125 logit scale)
    dim3 ggrid(WB * WS / 16, WE / 64);
    gemm_xwt_kernel<<<ggrid, 32, 0, stream>>>(Xqb, Wqb, bq, nullptr, Qb,  1, 0.125f);
    gemm_xwt_kernel<<<ggrid, 32, 0, stream>>>(Xkb, Wkb, bk, nullptr, Kb,  1, 1.0f);
    gemm_xwt_kernel<<<ggrid, 32, 0, stream>>>(Xvb, Wvb, bv, nullptr, Vbp, 1, 1.0f);

    // 3) V -> V^T [B,H,DK,S]
    vtrans_kernel<<<(NE + 255) / 256, 256, 0, stream>>>(Vbp, VTb);

    // 4) softmax stats
    attn_stats_kernel<<<WB * WH * (WS / 16), 32, 0, stream>>>(Qb, Kb, mstat, lstat);

    // 5) probs / threshold / adjacency / hidden
    size_t shmem = (size_t)16 * WS * sizeof(float) + (size_t)16 * 16 * 32 * sizeof(__bf16);
    attn_pass2_kernel<<<WB * (WS / 16), 512, shmem, stream>>>(
        Qb, Kb, VTb, mstat, lstat, Mb, adjp);

    // 6) out = merged @ Wp^T + bp  (fp32 to d_out)
    gemm_xwt_kernel<<<ggrid, 32, 0, stream>>>(Mb, Wpb, bp, outp, nullptr, 0, 1.0f);

    (void)in_sizes; (void)n_in; (void)out_size; (void)ws_size;
}
